// BalancedHamiltonLayer_61959198212534
// MI455X (gfx1250) — matrix operand order repaired and checked by reading, not compiled, not run on hardware
//
#include <hip/hip_runtime.h>
#include <stddef.h>

// ---------------------------------------------------------------------------
// BalancedHamiltonLayer on MI455X (gfx1250, wave32, WMMA)
//
// Reference collapses to a single GEMM:
//   out(8192x1024) = X(8192x1024) @ W^T(1024x1024) + bias
//   W[cout=k*64+j, cin=s*64+i] = sum_r H[r,k,s] * factors_B[r,j,i]
// Precision: fp32 emulated via bf16 hi/lo split (hi*hi + hi*lo + lo*hi),
// accumulated in f32 by v_wmma_f32_16x16x32_bf16.
//
// Block tile 128x256, 8 waves, wave tile 64x64 (4x4 WMMA accumulators):
//   per k-step/wave: 32 ds_load_b128 feed 48 WMMAs (10.7 B LDS per WMMA).
// Data movement:
//   X  : global fp32 -> VGPR -> (convert hi/lo bf16) -> LDS
//   W  : global bf16 -> LDS directly via GLOBAL_LOAD_ASYNC_TO_LDS_B128
//        (ASYNCcnt + s_wait_asynccnt), zero VGPR cost.
// ---------------------------------------------------------------------------

#define M_TOT 8192
#define K_TOT 1024
#define N_TOT 1024

#define BM 128
#define BN 256
#define BK 32
#define NSTEP (K_TOT / BK)   // 32 k-steps

#if defined(__has_builtin)
# if __has_builtin(__builtin_amdgcn_global_load_async_to_lds_b128) && \
     __has_builtin(__builtin_amdgcn_s_wait_asynccnt)
#  define USE_ASYNC_LDS 1
# endif
#endif
#ifndef USE_ASYNC_LDS
# define USE_ASYNC_LDS 0
#endif

typedef __attribute__((ext_vector_type(16))) __bf16          v16bf;
typedef __attribute__((ext_vector_type(8)))  float           v8f;
typedef __attribute__((ext_vector_type(4)))  float           v4f;
typedef __attribute__((ext_vector_type(8)))  unsigned short  v8us;
typedef __attribute__((ext_vector_type(16))) unsigned short  v16us;

__device__ __forceinline__ unsigned short f2bf(float f) {
  unsigned int u = __float_as_uint(f);
  u += 0x7fffu + ((u >> 16) & 1u);      // round-to-nearest-even
  return (unsigned short)(u >> 16);
}
__device__ __forceinline__ float bf2f(unsigned short h) {
  return __uint_as_float(((unsigned int)h) << 16);
}

// Two 16B LDS chunks -> one v16bf WMMA fragment
__device__ __forceinline__ v16bf ldfrag2(const unsigned short* p0,
                                         const unsigned short* p1) {
  v8us a = *(const v8us*)p0;
  v8us b = *(const v8us*)p1;
  v16us w = __builtin_shufflevector(a, b, 0,1,2,3,4,5,6,7,8,9,10,11,12,13,14,15);
  union { v16us u; v16bf f; } cv; cv.u = w;
  return cv.f;
}

// XOR swizzle: element offset of an 8-elem chunk inside a [rows x 32] bf16 tile,
// chosen so 16 lanes reading consecutive rows hit distinct bank quartets.
__device__ __forceinline__ int sw(int row, int c8) {
  return row * BK + (c8 ^ (((row >> 2) & 3) << 3));
}

#if USE_ASYNC_LDS
// Builtin signature (from hipcc diagnostic): param0 is a pointer to a 16-byte
// int vector in the global (AS1) address space; param1 is the LDS (AS3) side.
typedef int v4i_t __attribute__((vector_size(16)));
typedef __attribute__((address_space(1))) v4i_t* as1_v4i_ptr;
typedef __attribute__((address_space(3))) v4i_t* as3_v4i_ptr;

__device__ __forceinline__ void async_b128(const unsigned short* g,
                                           unsigned short* l) {
  __builtin_amdgcn_global_load_async_to_lds_b128(
      (as1_v4i_ptr)(g), (as3_v4i_ptr)(l), 0, 0);
}
#endif

// ------------------------- W precompute (tiny) ------------------------------
__device__ __constant__ int   c_comp[16] = {0,1,2,3, 1,0,3,2, 2,3,0,1, 3,2,1,0};
__device__ __constant__ float c_sign[16] = {1.f,-1.f,-1.f,-1.f,
                                            1.f, 1.f,-1.f, 1.f,
                                            1.f, 1.f, 1.f,-1.f,
                                            1.f,-1.f, 1.f, 1.f};

__global__ void build_w_kernel(const float* __restrict__ A,     // (8,4,4,4)
                               const float* __restrict__ fB,    // (8,64,64)
                               unsigned short* __restrict__ Wh, // (1024,1024) bf16
                               unsigned short* __restrict__ Wl) {
  const int idx  = blockIdx.x * 256 + threadIdx.x;   // 0 .. 1M-1
  const int cout = idx >> 10, cin = idx & 1023;
  const int k = cout >> 6, j = cout & 63;
  const int s = cin  >> 6, i = cin  & 63;
  const int bi = k >> 2, u = k & 3, bj = s >> 2, v = s & 3;
  const int   comp = c_comp[bi * 4 + bj];
  const float sg   = c_sign[bi * 4 + bj];
  float acc = 0.f;
#pragma unroll
  for (int r = 0; r < 8; ++r) {
    const float h = sg * A[((r * 4 + comp) * 4 + u) * 4 + v];
    acc += h * fB[(r * 64 + j) * 64 + i];
  }
  const unsigned short hi = f2bf(acc);
  Wh[idx] = hi;
  Wl[idx] = f2bf(acc - bf2f(hi));
}

// ------------------------------- GEMM ---------------------------------------
__global__ __launch_bounds__(256) void hamilton_gemm(
    const float*          __restrict__ X,
    const unsigned short* __restrict__ Wh,
    const unsigned short* __restrict__ Wl,
    const float*          __restrict__ bias,
    float*                __restrict__ out) {
  // 96 KB static LDS: double-buffered X (128x32) and W (256x32) hi/lo tiles
  __shared__ __align__(16) unsigned short sXh[2][BM * BK];
  __shared__ __align__(16) unsigned short sXl[2][BM * BK];
  __shared__ __align__(16) unsigned short sWh[2][BN * BK];
  __shared__ __align__(16) unsigned short sWl[2][BN * BK];

  const int tid   = threadIdx.x;
  const int lane  = tid & 31;
  const int wid   = tid >> 5;      // 8 waves
  const int waveM = wid & 1;       // 2 along M (64 rows each)
  const int waveN = wid >> 1;      // 4 along N (64 cols each)

  const int bm = blockIdx.x * BM;
  const int bn = blockIdx.y * BN;

  // --- X staging: each thread owns a 16-wide run of one of 128 rows ---
  const int srow = tid >> 1;               // 0..127
  const int scol = (tid & 1) << 4;         // 0 or 16
  const float* xg = X + (size_t)(bm + srow) * K_TOT + scol;
  const int so0 = sw(srow, scol);          // swizzled chunk offsets
  const int so1 = sw(srow, scol + 8);

  // --- W staging: each thread owns one of 256 rows (4 chunks of 8) ---
  const unsigned short* whg = Wh + (size_t)(bn + tid) * K_TOT;
  const unsigned short* wlg = Wl + (size_t)(bn + tid) * K_TOT;
  int swW[4];
#pragma unroll
  for (int c = 0; c < 4; ++c) swW[c] = sw(tid, c * 8);

  v4f xr0, xr1, xr2, xr3;

  auto load_x = [&](int ks) {
    const float* xp = xg + ks * BK;
    xr0 = *(const v4f*)(xp + 0);
    xr1 = *(const v4f*)(xp + 4);
    xr2 = *(const v4f*)(xp + 8);
    xr3 = *(const v4f*)(xp + 12);
  };

#if USE_ASYNC_LDS
  auto issue_w = [&](int ks, int buf) {
#pragma unroll
    for (int c = 0; c < 4; ++c) {
      const int c8 = c * 8;
      async_b128(whg + ks * BK + c8, &sWh[buf][swW[c]]);
      async_b128(wlg + ks * BK + c8, &sWl[buf][swW[c]]);
    }
  };
#else
  auto copy_w = [&](int ks, int buf) {     // fallback: sync copy at store point
#pragma unroll
    for (int c = 0; c < 4; ++c) {
      const int c8 = c * 8;
      *(v8us*)&sWh[buf][swW[c]] = *(const v8us*)(whg + ks * BK + c8);
      *(v8us*)&sWl[buf][swW[c]] = *(const v8us*)(wlg + ks * BK + c8);
    }
  };
#endif

  auto store_x = [&](int buf) {
    float xv[16];
#pragma unroll
    for (int c = 0; c < 4; ++c) {
      xv[c + 0]  = xr0[c];
      xv[c + 4]  = xr1[c];
      xv[c + 8]  = xr2[c];
      xv[c + 12] = xr3[c];
    }
    v8us h0, h1, l0, l1;
#pragma unroll
    for (int c = 0; c < 8; ++c) {
      const unsigned short ha = f2bf(xv[c]);
      const unsigned short hb = f2bf(xv[c + 8]);
      h0[c] = ha;  l0[c] = f2bf(xv[c]     - bf2f(ha));
      h1[c] = hb;  l1[c] = f2bf(xv[c + 8] - bf2f(hb));
    }
    *(v8us*)&sXh[buf][so0] = h0;  *(v8us*)&sXh[buf][so1] = h1;
    *(v8us*)&sXl[buf][so0] = l0;  *(v8us*)&sXl[buf][so1] = l1;
  };

  v8f acc[4][4];
  const v8f vzero = {0.f,0.f,0.f,0.f,0.f,0.f,0.f,0.f};
#pragma unroll
  for (int mi = 0; mi < 4; ++mi)
#pragma unroll
    for (int ni = 0; ni < 4; ++ni) acc[mi][ni] = vzero;

  // Fragment addressing per ISA VGPR layouts (wave32):
  //  A 16x32 bf16: lane%16 = M row, lane/16 selects K-halves {0-7,16-23} / {8-15,24-31}
  //  B 32x16 bf16: lane%16 = N col, lane/16 selects K-halves {0-7,16-23} / {8-15,24-31}
  const int arow = waveM * 64 + (lane & 15);
  const int akb  = (lane >> 4) << 3;   // 0 / 8
  const int brow = waveN * 64 + (lane & 15);
  const int bkb  = (lane >> 4) << 3;   // 0 / 8

  // ---- prologue: fill buffer 0 ----
#if USE_ASYNC_LDS
  issue_w(0, 0);
#endif
  load_x(0);
  store_x(0);
#if USE_ASYNC_LDS
  __builtin_amdgcn_s_wait_asynccnt(0);
#else
  copy_w(0, 0);
#endif
  __syncthreads();

  for (int ks = 0; ks < NSTEP; ++ks) {
    const int cur = ks & 1;
    if (ks + 1 < NSTEP) {
      // buf cur^1 is free: all waves passed the barrier that ended step ks-1
#if USE_ASYNC_LDS
      issue_w(ks + 1, cur ^ 1);       // global -> LDS, no VGPR bounce
#endif
      load_x(ks + 1);                 // reg-level global prefetch (fp32 X)
    }
    if (ks + 2 < NSTEP) {             // L2-level prefetch (+2 steps)
      __builtin_prefetch(xg  + (ks + 2) * BK, 0, 0);
      __builtin_prefetch(whg + (ks + 2) * BK, 0, 0);
      __builtin_prefetch(wlg + (ks + 2) * BK, 0, 0);
    }

    v16bf ah[4], al[4];
#pragma unroll
    for (int mi = 0; mi < 4; ++mi) {
      const int r = arow + mi * 16;
      ah[mi] = ldfrag2(&sXh[cur][sw(r, akb)], &sXh[cur][sw(r, akb + 16)]);
      al[mi] = ldfrag2(&sXl[cur][sw(r, akb)], &sXl[cur][sw(r, akb + 16)]);
    }
#pragma unroll
    for (int ni = 0; ni < 4; ++ni) {
      const int r = brow + ni * 16;
      const v16bf bh = ldfrag2(&sWh[cur][sw(r, bkb)], &sWh[cur][sw(r, bkb + 16)]);
      const v16bf bl = ldfrag2(&sWl[cur][sw(r, bkb)], &sWl[cur][sw(r, bkb + 16)]);
#pragma unroll
      for (int mi = 0; mi < 4; ++mi) {
        acc[mi][ni] = __builtin_amdgcn_wmma_f32_16x16x32_bf16(
            false, ah[mi], false, bh, (short)0, acc[mi][ni], false, false);
        acc[mi][ni] = __builtin_amdgcn_wmma_f32_16x16x32_bf16(
            false, ah[mi], false, bl, (short)0, acc[mi][ni], false, false);
        acc[mi][ni] = __builtin_amdgcn_wmma_f32_16x16x32_bf16(
            false, al[mi], false, bh, (short)0, acc[mi][ni], false, false);
      }
    }
    __syncthreads();                  // all waves done reading buf cur
    if (ks + 1 < NSTEP) {
      store_x(cur ^ 1);
#if USE_ASYNC_LDS
      __builtin_amdgcn_s_wait_asynccnt(0);   // this wave's W tiles landed
#else
      copy_w(ks + 1, cur ^ 1);
#endif
      __syncthreads();                // W + X of buf cur^1 visible to all
    }
  }

  // Epilogue. C/D layout: VGPR v, lanes 0-15 -> M=v, lanes 16-31 -> M=v+8; N=lane%16
  const int r0 = bm + waveM * 64 + ((lane >> 4) << 3);
  const int c0 = bn + waveN * 64 + (lane & 15);
#pragma unroll
  for (int ni = 0; ni < 4; ++ni) {
    const float bv = bias[c0 + ni * 16];
#pragma unroll
    for (int mi = 0; mi < 4; ++mi) {
      const v8f a = acc[mi][ni];
#pragma unroll
      for (int v = 0; v < 8; ++v) {
        out[(size_t)(r0 + mi * 16 + v) * N_TOT + (c0 + ni * 16)] = a[v] + bv;
      }
    }
  }
}

// ------------------------------ launch --------------------------------------
extern "C" void kernel_launch(void* const* d_in, const int* in_sizes, int n_in,
                              void* d_out, int out_size, void* d_ws, size_t ws_size,
                              hipStream_t stream) {
  const float* x    = (const float*)d_in[0];   // (4,2048,1024) fp32
  const float* A    = (const float*)d_in[1];   // (8,4,4,4)     fp32
  const float* fB   = (const float*)d_in[2];   // (8,64,64)     fp32
  const float* bias = (const float*)d_in[3];   // (1024,)       fp32
  float*       out  = (float*)d_out;           // (4,2048,1024) fp32

  unsigned short* Wh = (unsigned short*)d_ws;              // 2 MB
  unsigned short* Wl = Wh + (size_t)N_TOT * K_TOT;         // 2 MB

  // 1) fuse H and factors_B into one 1024x1024 weight, split hi/lo bf16
  build_w_kernel<<<(N_TOT * K_TOT) / 256, 256, 0, stream>>>(A, fB, Wh, Wl);

  // 2) out = X @ W^T + bias  via split-bf16 WMMA
  dim3 grid(M_TOT / BM, N_TOT / BN);   // 64 x 4
  hamilton_gemm<<<grid, 256, 0, stream>>>(x, Wh, Wl, bias, out);
}
